// HVAE_61392262529344
// MI455X (gfx1250) — matrix operand order repaired and checked by reading, not compiled, not run on hardware
//
#include <hip/hip_runtime.h>
#include <hip/hip_bf16.h>
#include <cstdint>

typedef __bf16 bf16_t;
typedef bf16_t bf16x8 __attribute__((ext_vector_type(8)));
typedef bf16_t v16bf  __attribute__((ext_vector_type(16)));
typedef float  v8f    __attribute__((ext_vector_type(8)));
typedef int    v4i_t  __attribute__((ext_vector_type(4)));

#define TDEPTH 6
#define BATCH  1024
#define HID    256
#define OUTV   32
#define BM     256
#define BN     64
#define BK     32
#define APITCH 40   // 32 + 8 halfs pad -> 80B row pitch, 16B aligned, conflict-free b128 reads

// CDNA5 async global->LDS path (ASYNCcnt-tracked), with register-staged fallback.
#if defined(__has_builtin)
#if __has_builtin(__builtin_amdgcn_global_load_async_to_lds_b128)
#define USE_ASYNC_LDS 1
#endif
#endif

// Builtin takes typed v4i pointers: AS1 (prints as __device__) src, AS3 (__shared__) dst.
// Route through uintptr_t so the AS conversion is an int-to-pointer cast (always legal);
// the 32-bit truncation for AS3 is exactly the ISA's flat->LDS address truncation.
#define AS1P(p) ((__attribute__((address_space(1))) v4i_t*)(uintptr_t)(p))
#define AS3P(p) ((__attribute__((address_space(3))) v4i_t*)(uint32_t)(uintptr_t)(p))

__device__ __forceinline__ void async_tile_wait() {
#ifdef USE_ASYNC_LDS
#if __has_builtin(__builtin_amdgcn_s_wait_asynccnt)
  __builtin_amdgcn_s_wait_asynccnt(0);
#else
  asm volatile("s_wait_asynccnt 0x0" ::: "memory");
#endif
#endif
}

// Preorder index of node at (level, pos) in a full binary tree of height TDEPTH.
__device__ __forceinline__ int preorder_idx(int level, int pos) {
  int idx = 0, hh = TDEPTH;
  for (int j = 0; j < level; ++j) {
    int bit = (pos >> (level - 1 - j)) & 1;
    idx += bit ? (1 << (hh - 1)) : 1;
    hh--;
  }
  return idx;
}

// C[map(row), n] = sum_k A[mapA(row), k] * W[n, k] + bias[n]
// A rows are (node, batch) pairs; node remapping: src_node = n*SA+TA, dst_node = n*SC+TC (+dup).
// predMode: scatter into d_out at preorder slot of pos = node*posScale+posOff (level childLevel).
__global__ __launch_bounds__(256)
void gemm_bf16_kernel(const bf16_t* __restrict__ A, int SA, int TA, int K,
                      const bf16_t* __restrict__ W,
                      const float* __restrict__ bias,
                      float* __restrict__ C, int ldc, int SC, int TC, int dup,
                      bf16_t* __restrict__ Cbf,
                      int predMode, int childLevel, int posScale, int posOff,
                      int N)
{
  __shared__ __align__(16) bf16_t sA[2][BM][APITCH];
  __shared__ __align__(16) bf16_t sB[2][BN][APITCH];

  const int tid  = threadIdx.x;
  const int m0   = blockIdx.y * BM;
  const int n0   = blockIdx.x * BN;
  const int lane = tid & 31;
  const int wv   = tid >> 5;            // 8 waves: 4 along M x 2 along N
  const int wvm  = (wv >> 1) * 64;      // wave tile: 64 x 32
  const int wvn  = (wv & 1) * 32;
  const int half = lane >> 4;
  const int lr   = lane & 15;

  v8f acc[4][2];
#pragma unroll
  for (int mi = 0; mi < 4; ++mi)
#pragma unroll
    for (int ni = 0; ni < 2; ++ni) {
      v8f zr = {0.f,0.f,0.f,0.f,0.f,0.f,0.f,0.f};
      acc[mi][ni] = zr;
    }

  auto loadTile = [&](int buf, int k0) {
    // A tile: BM x BK halfs, 16B chunks: BM*4 = 1024 chunks over 256 threads
#pragma unroll
    for (int it = 0; it < 4; ++it) {
      int i = tid + it * 256;
      int r = i >> 2, c = (i & 3) * 8;
      int grow = m0 + r;
      int nn = grow >> 10, rb = grow & 1023;
      const bf16_t* src = A + ((size_t)(nn * SA + TA) * BATCH + rb) * (size_t)K + k0 + c;
#ifdef USE_ASYNC_LDS
      __builtin_amdgcn_global_load_async_to_lds_b128(AS1P(src), AS3P(&sA[buf][r][c]), 0, 0);
#else
      *(bf16x8*)&sA[buf][r][c] = *(const bf16x8*)src;
#endif
    }
    // W tile: BN x BK halfs, 256 chunks over 256 threads (over-read past N is in-workspace)
    {
      int r = tid >> 2, c = (tid & 3) * 8;
      const bf16_t* src = W + (size_t)(n0 + r) * (size_t)K + k0 + c;
#ifdef USE_ASYNC_LDS
      __builtin_amdgcn_global_load_async_to_lds_b128(AS1P(src), AS3P(&sB[buf][r][c]), 0, 0);
#else
      *(bf16x8*)&sB[buf][r][c] = *(const bf16x8*)src;
#endif
    }
  };

  loadTile(0, 0);
  const int KT = K / BK;
  for (int kt = 0; kt < KT; ++kt) {
    int cur = kt & 1;
    async_tile_wait();
    __syncthreads();
    if (kt + 1 < KT) loadTile(cur ^ 1, (kt + 1) * BK);

    // A fragment: lane<16 -> row, K {0-7,16-23}; lane>=16 -> row, K {8-15,24-31}
    v16bf af[4], bfr[2];
#pragma unroll
    for (int mi = 0; mi < 4; ++mi) {
      int row = wvm + mi * 16 + lr;
      bf16x8 lo = *(const bf16x8*)&sA[cur][row][half * 8];
      bf16x8 hi = *(const bf16x8*)&sA[cur][row][half * 8 + 16];
      af[mi] = __builtin_shufflevector(lo, hi, 0,1,2,3,4,5,6,7,8,9,10,11,12,13,14,15);
    }
    // B fragment: lane<16 -> K {0-7,16-23}; lane>=16 -> K {8-15,24-31}, lane%16 = column n; B[k,n] = W[n,k]
#pragma unroll
    for (int ni = 0; ni < 2; ++ni) {
      int col = wvn + ni * 16 + lr;
      bf16x8 lo = *(const bf16x8*)&sB[cur][col][half * 8];
      bf16x8 hi = *(const bf16x8*)&sB[cur][col][half * 8 + 16];
      bfr[ni] = __builtin_shufflevector(lo, hi, 0,1,2,3,4,5,6,7,8,9,10,11,12,13,14,15);
    }
#pragma unroll
    for (int mi = 0; mi < 4; ++mi)
#pragma unroll
      for (int ni = 0; ni < 2; ++ni)
        acc[mi][ni] = __builtin_amdgcn_wmma_f32_16x16x32_bf16(
            false, af[mi], false, bfr[ni], (short)0, acc[mi][ni], false, false);
  }

  // Epilogue: D layout lanes0-15 M=v, lanes16-31 M=v+8; N = lane%16
#pragma unroll
  for (int ni = 0; ni < 2; ++ni) {
    int col = n0 + wvn + ni * 16 + lr;
    if (col >= N) continue;            // only pred GEMMs (N=32) hit this
    float bv = bias ? bias[col] : 0.0f;
#pragma unroll
    for (int mi = 0; mi < 4; ++mi) {
#pragma unroll
      for (int v = 0; v < 8; ++v) {
        int grow = m0 + wvm + mi * 16 + half * 8 + v;
        int nn = grow >> 10, rb = grow & 1023;
        float val = acc[mi][ni][v] + bv;
        if (predMode) {
          int pos = nn * posScale + posOff;
          int pre = preorder_idx(childLevel, pos);
          C[((size_t)pre * BATCH + rb) * (size_t)ldc + col] = val;
        } else {
          size_t dst = ((size_t)(nn * SC + TC) * BATCH + rb) * (size_t)ldc + col;
          C[dst] = val;
          if (dup) C[dst + (size_t)BATCH * ldc] = val;
          if (Cbf) {
            Cbf[dst] = (bf16_t)val;
            if (dup) Cbf[dst + (size_t)BATCH * ldc] = (bf16_t)val;
          }
        }
      }
    }
  }
}

// One wave32 per row: 32 lanes == 32 logits. Reads preds from d_out (preorder slot),
// writes softmax (bf16) at node position pos = node*posScale+posOff.
__global__ __launch_bounds__(256)
void softmax_kernel(const float* __restrict__ pred, bf16_t* __restrict__ spbf,
                    int childLevel, int posScale, int posOff)
{
  int row  = blockIdx.x * 8 + (threadIdx.x >> 5);
  int lane = threadIdx.x & 31;
  int nn = row >> 10, rb = row & 1023;
  int pos = nn * posScale + posOff;
  int pre = preorder_idx(childLevel, pos);
  float x = pred[((size_t)pre * BATCH + rb) * OUTV + lane];
  float m = x;
  for (int o = 16; o; o >>= 1) m = fmaxf(m, __shfl_xor(m, o, 32));
  float e = __expf(x - m);
  float s = e;
  for (int o = 16; o; o >>= 1) s += __shfl_xor(s, o, 32);
  spbf[((size_t)pos * BATCH + rb) * OUTV + lane] = (bf16_t)(e / s);
}

// GRU combine: Gx/Gh are [M,768] = gate-major pre-activations (biases already added).
// r = sig(Gx0+Gh0); z = sig(Gx1+Gh1); n = tanh(Gx2 + r*Gh2); out = (1-z)*n + z*h (+addSrc)
__global__ __launch_bounds__(256)
void gru_combine_kernel(const float* __restrict__ Gx, const float* __restrict__ Gh,
                        const float* __restrict__ hsrc, int hS, int hT,
                        const float* __restrict__ addSrc,
                        float* __restrict__ outF, bf16_t* __restrict__ outBf)
{
  size_t idx = (size_t)blockIdx.x * 256 + threadIdx.x;
  size_t row = idx >> 8;
  int j  = (int)(idx & 255);
  int nn = (int)(row >> 10), rb = (int)(row & 1023);
  size_t g = row * 768 + j;
  float gx0 = Gx[g], gx1 = Gx[g + 256], gx2 = Gx[g + 512];
  float gh0 = Gh[g], gh1 = Gh[g + 256], gh2 = Gh[g + 512];
  float h = hsrc[((size_t)(nn * hS + hT) * BATCH + rb) * HID + j];
  float r  = 1.0f / (1.0f + __expf(-(gx0 + gh0)));
  float zz = 1.0f / (1.0f + __expf(-(gx1 + gh1)));
  float n  = tanhf(gx2 + r * gh2);
  float v  = (1.0f - zz) * n + zz * h;
  if (addSrc) v += addSrc[idx];
  if (outF)  outF[idx]  = v;
  if (outBf) outBf[idx] = (bf16_t)v;
}

__global__ __launch_bounds__(256)
void cvt_kernel(const float* __restrict__ in, bf16_t* __restrict__ out, int n)
{
  int i = blockIdx.x * 256 + threadIdx.x;
  if (i < n) out[i] = (bf16_t)in[i];
}

extern "C" void kernel_launch(void* const* d_in, const int* in_sizes, int n_in,
                              void* d_out, int out_size, void* d_ws, size_t ws_size,
                              hipStream_t stream)
{
  (void)in_sizes; (void)n_in; (void)out_size; (void)ws_size;
  const float* z     = (const float*)d_in[0];
  const float* h2o_b = (const float*)d_in[2];
  const float* wc_b  = (const float*)d_in[4];
  const float* wd_b  = (const float*)d_in[6];
  const float* z1_b  = (const float*)d_in[8];
  const float* z2_b  = (const float*)d_in[10];
  const float* ga_bx = (const float*)d_in[13];
  const float* ga_bh = (const float*)d_in[14];
  const float* gs_bx = (const float*)d_in[17];
  const float* gs_bh = (const float*)d_in[18];

  char* ws = (char*)d_ws;
  size_t off = 0;
  auto carve = [&](size_t bytes) -> char* {
    char* p = ws + off;
    off = (off + bytes + 255) & ~(size_t)255;
    return p;
  };
  const size_t RMAX = 32u * 1024u;  // rows at deepest child level (32 nodes * 1024)
  const size_t RC   = 16u * 1024u;  // rows at deepest processed level

  float*  haF[2] = { (float*)carve(RMAX*HID*4), (float*)carve(RMAX*HID*4) };
  float*  hF     =   (float*)carve(RMAX*HID*4);                 // this-level children f32
  bf16_t* haB[2] = { (bf16_t*)carve(RMAX*HID*2), (bf16_t*)carve(RMAX*HID*2) };
  bf16_t* hB[2]  = { (bf16_t*)carve(RMAX*HID*2), (bf16_t*)carve(RMAX*HID*2) };
  bf16_t* spB[2] = { (bf16_t*)carve(RMAX*OUTV*2), (bf16_t*)carve(RMAX*OUTV*2) };
  float*  Gx  = (float*)carve(RC*768*4);
  float*  Gh  = (float*)carve(RC*768*4);
  float*  t1  = (float*)carve(RC*HID*4);
  bf16_t* h2B = (bf16_t*)carve(RC*HID*2);
  bf16_t* zB    = (bf16_t*)carve((size_t)BATCH*HID*2);
  bf16_t* h2oB  = (bf16_t*)carve((size_t)OUTV*HID*2);
  bf16_t* wcB   = (bf16_t*)carve((size_t)HID*HID*2);
  bf16_t* wdB   = (bf16_t*)carve((size_t)HID*HID*2);
  bf16_t* z1B   = (bf16_t*)carve((size_t)HID*HID*2);
  bf16_t* z2B   = (bf16_t*)carve((size_t)HID*HID*2);
  bf16_t* gawxB = (bf16_t*)carve((size_t)768*OUTV*2);
  bf16_t* gawhB = (bf16_t*)carve((size_t)768*HID*2);
  bf16_t* gswxB = (bf16_t*)carve((size_t)768*OUTV*2);
  bf16_t* gswhB = (bf16_t*)carve((size_t)768*HID*2);

  auto cvt = [&](const float* in, bf16_t* outp, int n) {
    cvt_kernel<<<(n + 255) / 256, 256, 0, stream>>>(in, outp, n);
  };
  cvt(z, zB, BATCH*HID);
  cvt((const float*)d_in[1],  h2oB,  OUTV*HID);
  cvt((const float*)d_in[3],  wcB,   HID*HID);
  cvt((const float*)d_in[5],  wdB,   HID*HID);
  cvt((const float*)d_in[7],  z1B,   HID*HID);
  cvt((const float*)d_in[9],  z2B,   HID*HID);
  cvt((const float*)d_in[11], gawxB, 768*OUTV);
  cvt((const float*)d_in[12], gawhB, 768*HID);
  cvt((const float*)d_in[15], gswxB, 768*OUTV);
  cvt((const float*)d_in[16], gswhB, 768*HID);

  auto gemm = [&](const bf16_t* A, int SA, int TA, int K,
                  const bf16_t* W, const float* bias,
                  float* C, int ldc, int SC, int TC, int dup, bf16_t* Cbf,
                  int predMode, int childLevel, int posScale, int posOff,
                  int M, int N) {
    dim3 grid((N + BN - 1) / BN, M / BM);
    gemm_bf16_kernel<<<grid, 256, 0, stream>>>(A, SA, TA, K, W, bias, C, ldc, SC, TC,
                                               dup, Cbf, predMode, childLevel,
                                               posScale, posOff, N);
  };

  float* dout = (float*)d_out;

  // ---- root init: ha0 = z@z2h1^T, h0 = z@z2h2^T, pred0, sp0 ----
  gemm(zB,1,0,HID,  z1B, z1_b, haF[0],HID,1,0,0, haB[0], 0,0,0,0, BATCH, HID);
  gemm(zB,1,0,HID,  z2B, z2_b, hF,    HID,1,0,0, hB[0],  0,0,0,0, BATCH, HID);
  gemm(hB[0],1,0,HID, h2oB, h2o_b, dout,OUTV,0,0,0, nullptr, 1,0,1,0, BATCH, OUTV);
  softmax_kernel<<<BATCH/8, 256, 0, stream>>>(dout, spB[0], 0, 1, 0);

  // ---- level-by-level (levels 0..4 have children) ----
  for (int L = 0; L < TDEPTH - 1; ++L) {
    int cur = L & 1, nxt = cur ^ 1;
    int nodes = 1 << L;
    int M = nodes << 10;   // nodes * BATCH rows
    int CL = L + 1;        // children's level

    // 1) c = ha@w_c^T + b  -> duplicated to both children's ancestral state
    gemm(haB[cur],1,0,HID, wcB, wc_b, haF[nxt],HID,2,0,1, haB[nxt], 0,0,0,0, M, HID);
    // 2) h_left = h@w_d^T + b
    gemm(hB[cur],1,0,HID,  wdB, wd_b, hF,HID,2,0,0, hB[nxt], 0,0,0,0, M, HID);
    // 3) pred_left -> d_out (preorder scatter), 4) softmax_left -> sp
    gemm(hB[nxt],2,0,HID, h2oB, h2o_b, dout,OUTV,0,0,0, nullptr, 1,CL,2,0, M, OUTV);
    softmax_kernel<<<M/8, 256, 0, stream>>>(dout, spB[nxt], CL, 2, 0);
    // 5/6) gru_s gate pre-activations (x-side K=32, h-side K=256), 7) combine -> t1
    gemm(spB[nxt],2,0,OUTV, gswxB, gs_bx, Gx,768,1,0,0, nullptr, 0,0,0,0, M, 768);
    gemm(hB[nxt],2,0,HID,   gswhB, gs_bh, Gh,768,1,0,0, nullptr, 0,0,0,0, M, 768);
    gru_combine_kernel<<<M, 256, 0, stream>>>(Gx, Gh, hF, 2, 0, nullptr, t1, nullptr);
    // 8/9) gru_a gates, 10) hidden2 = t1 + gru_a(...) -> bf16
    gemm(spB[cur],1,0,OUTV, gawxB, ga_bx, Gx,768,1,0,0, nullptr, 0,0,0,0, M, 768);
    gemm(haB[cur],1,0,HID,  gawhB, ga_bh, Gh,768,1,0,0, nullptr, 0,0,0,0, M, 768);
    gru_combine_kernel<<<M, 256, 0, stream>>>(Gx, Gh, haF[cur], 1, 0, t1, nullptr, h2B);
    // 11) h_right = hidden2@w_d^T + b
    gemm(h2B,1,0,HID, wdB, wd_b, hF,HID,2,1,0, hB[nxt], 0,0,0,0, M, HID);
    // 12/13) pred_right + softmax_right
    gemm(hB[nxt],2,1,HID, h2oB, h2o_b, dout,OUTV,0,0,0, nullptr, 1,CL,2,1, M, OUTV);
    softmax_kernel<<<M/8, 256, 0, stream>>>(dout, spB[nxt], CL, 2, 1);
  }
}
